// SubmanifoldSparseConv_730144440356
// MI455X (gfx1250) — matrix operand order repaired and checked by reading, not compiled, not run on hardware
//
#include <hip/hip_runtime.h>

// Submanifold sparse 3^3 conv, C_in=C_out=32, via v_wmma_f32_16x16x32_f16.
// One wave32 = one tile of 16 points; accumulator = two 16x16 f32 C tiles.
// Software-pipelined neighbor gather; non-temporal output stores keep the
// feature tensor + voxel grid resident in the 192MB L2.

#define GRID_E 160
#define GP     162                 // padded grid extent
#define G3     (GP * GP * GP)      // 4,251,528 voxels
#define CIN    32
#define COUT   32
#define K3     27

typedef __attribute__((ext_vector_type(16))) _Float16 v16h;
typedef __attribute__((ext_vector_type(8)))  float    v8f;

__global__ void fill_grid_kernel(int* __restrict__ g, int total) {
    int i = blockIdx.x * blockDim.x + threadIdx.x;
    if (i < total) g[i] = -1;
}

__global__ void scatter_grid_kernel(const int* __restrict__ pos,
                                    int* __restrict__ g, int N) {
    int i = blockIdx.x * blockDim.x + threadIdx.x;
    int b = blockIdx.y;
    if (i >= N) return;
    const int* p = pos + ((size_t)b * N + i) * 3;
    int lin = ((p[0] + 1) * GP + (p[1] + 1)) * GP + (p[2] + 1);
    g[(size_t)b * G3 + lin] = i;
}

// Pack W[27][32][32] (f32, Cin-major) into the exact 16-bit B-matrix (32x16)
// lane layout: entry (k, ntile, lane) holds 16 f16 = column (ntile*16 + lane%16)
// of W[k], K rows (lane<16 ? {0..7,16..23} : {8..15,24..31}), contiguous 32 bytes per lane.
__global__ void prepack_w_kernel(const float* __restrict__ W,
                                 _Float16* __restrict__ pw) {
    int t = blockIdx.x * blockDim.x + threadIdx.x;   // [0, 27*2*32)
    if (t >= K3 * 2 * 32) return;
    int lane = t & 31;
    int k    = t >> 6;
    int col   = ((t >> 5) & 1) * 16 + (lane & 15);
    int kbase = (lane < 16) ? 0 : 8;
    _Float16* dst = pw + (size_t)t * 16;
#pragma unroll
    for (int j = 0; j < 16; ++j)
        dst[j] = (_Float16)W[((size_t)k * CIN + (kbase + j + (j >= 8 ? 8 : 0))) * COUT + col];
}

__device__ __forceinline__ int neigh_off(int k) {
    int dx = k / 9 - 1, dy = (k / 3) % 3 - 1, dz = k % 3 - 1;
    return (dx * GP + dy) * GP + dz;
}

__global__ void __launch_bounds__(256)
sconv_wmma_kernel(const float* __restrict__ feat,
                  const int*   __restrict__ pos,
                  const int*   __restrict__ grid,
                  const _Float16* __restrict__ pw,
                  float* __restrict__ out,
                  int N, int ntiles) {
    int tile = blockIdx.x * (blockDim.x >> 5) + (threadIdx.x >> 5);
    if (tile >= ntiles) return;                    // wave-uniform exit
    int b    = blockIdx.y;
    int lane = threadIdx.x & 31;
    int m    = lane & 15;                          // matrix row this lane feeds
    int row  = tile * 16 + m;                      // point index
    bool row_ok = row < N;

    int lin = 0;
    if (row_ok) {
        const int* p = pos + ((size_t)b * N + row) * 3;
        lin = ((p[0] + 1) * GP + (p[1] + 1)) * GP + (p[2] + 1);
    }
    const int*   gb = grid + (size_t)b * G3;
    const float* fb = feat + (size_t)b * N * CIN;
    // 16-bit A layout: lanes 0-15 own K {0..7,16..23}; lanes 16-31 own {8..15,24..31}
    int c0 = (lane < 16) ? 0 : 8;

    v8f acc0 = {};   // out channels 0..15
    v8f acc1 = {};   // out channels 16..31

    const float4 zero4 = make_float4(0.f, 0.f, 0.f, 0.f);

    // ---- software pipeline: stage 0 gather ----
    float4 f0 = zero4, f1 = zero4, f2 = zero4, f3 = zero4;
    {
        int idx = row_ok ? gb[lin + neigh_off(0)] : -1;
        if (idx >= 0) {
            const float* fr = fb + (size_t)idx * CIN;
            f0 = *(const float4*)(fr + c0);
            f1 = *(const float4*)(fr + c0 + 4);
            f2 = *(const float4*)(fr + c0 + 16);
            f3 = *(const float4*)(fr + c0 + 20);
        }
    }

    for (int k = 0; k < K3; ++k) {
        // issue next iteration's gather before consuming current data
        float4 g0 = zero4, g1 = zero4, g2 = zero4, g3 = zero4;
        if (k + 1 < K3) {
            int idn = row_ok ? gb[lin + neigh_off(k + 1)] : -1;
            if (idn >= 0) {
                const float* fr = fb + (size_t)idn * CIN;
                g0 = *(const float4*)(fr + c0);
                g1 = *(const float4*)(fr + c0 + 4);
                g2 = *(const float4*)(fr + c0 + 16);
                g3 = *(const float4*)(fr + c0 + 20);
            }
        }

        v16h a;
        a[0]  = (_Float16)f0.x; a[1]  = (_Float16)f0.y;
        a[2]  = (_Float16)f0.z; a[3]  = (_Float16)f0.w;
        a[4]  = (_Float16)f1.x; a[5]  = (_Float16)f1.y;
        a[6]  = (_Float16)f1.z; a[7]  = (_Float16)f1.w;
        a[8]  = (_Float16)f2.x; a[9]  = (_Float16)f2.y;
        a[10] = (_Float16)f2.z; a[11] = (_Float16)f2.w;
        a[12] = (_Float16)f3.x; a[13] = (_Float16)f3.y;
        a[14] = (_Float16)f3.z; a[15] = (_Float16)f3.w;

        v16h b0 = *(const v16h*)(pw + ((size_t)(k * 2 + 0) * 32 + lane) * 16);
        v16h b1 = *(const v16h*)(pw + ((size_t)(k * 2 + 1) * 32 + lane) * 16);

        acc0 = __builtin_amdgcn_wmma_f32_16x16x32_f16(
            false, a, false, b0, (short)0, acc0, false, false);
        acc1 = __builtin_amdgcn_wmma_f32_16x16x32_f16(
            false, a, false, b1, (short)0, acc1, false, false);

        f0 = g0; f1 = g1; f2 = g2; f3 = g3;
    }

    // C/D layout: VGPR r -> M=r (lanes 0-15) / M=8+r (lanes 16-31), N=lane&15
#pragma unroll
    for (int r = 0; r < 8; ++r) {
        int mrow = (lane < 16) ? r : (8 + r);
        int orow = tile * 16 + mrow;
        if (orow < N) {
            float* o = out + ((size_t)b * N + orow) * COUT + (lane & 15);
            __builtin_nontemporal_store(acc0[r], o);        // keep L2 for feat
            __builtin_nontemporal_store(acc1[r], o + 16);
        }
    }
}

extern "C" void kernel_launch(void* const* d_in, const int* in_sizes, int n_in,
                              void* d_out, int out_size, void* d_ws, size_t ws_size,
                              hipStream_t stream) {
    const float* feat = (const float*)d_in[0];   // [B, N, 32] f32
    const int*   pos  = (const int*)d_in[1];     // [B, N, 3]  i32
    const float* W    = (const float*)d_in[2];   // [27, 32, 32] f32
    float* out = (float*)d_out;

    const int B = 2;
    const int N = in_sizes[1] / (3 * B);

    // Scratch layout: [B * G3 ints grid hash][packed f16 W 27*2*32*16]
    int*       grid = (int*)d_ws;
    _Float16*  pw   = (_Float16*)((char*)d_ws + (size_t)B * G3 * sizeof(int));

    int totalg = B * G3;
    fill_grid_kernel<<<(totalg + 255) / 256, 256, 0, stream>>>(grid, totalg);

    dim3 sgrid((N + 255) / 256, B);
    scatter_grid_kernel<<<sgrid, 256, 0, stream>>>(pos, grid, N);

    prepack_w_kernel<<<(K3 * 2 * 32 + 255) / 256, 256, 0, stream>>>(W, pw);

    int ntiles = (N + 15) / 16;
    int wpb    = 256 / 32;                       // 8 waves per block
    dim3 cgrid((ntiles + wpb - 1) / wpb, B);
    sconv_wmma_kernel<<<cgrid, 256, 0, stream>>>(feat, pos, grid, pw, out,
                                                 N, ntiles);
}
